// SSMBlock_6270652252576
// MI455X (gfx1250) — matrix operand order repaired and checked by reading, not compiled, not run on hardware
//
#include <hip/hip_runtime.h>
#include <hip/hip_bf16.h>

typedef _Float16 v16h __attribute__((ext_vector_type(16)));
typedef _Float16 h8   __attribute__((ext_vector_type(8)));
typedef float    v8f  __attribute__((ext_vector_type(8)));

#define LSEQ    64
#define DM      512
#define DI      1024
#define DS      16
#define DTRANK  32

// ---------------------------------------------------------------------------
// Kernel 0: AvgPool3d(16,16,16) over (64,128,128,128) -> seqH f16 (64 x 512)
// One 256-thread block per (channel, z-block, y-block) strip = 128 KB of
// contiguous-per-plane data. Each thread's float4 always lands in the same
// x-bucket, so accumulation is a single register; tiny LDS reduction at end.
// Pure streaming: this is the bandwidth-bound 512 MB read.
// ---------------------------------------------------------------------------
__global__ void pool_kernel(const float* __restrict__ x, _Float16* __restrict__ seqH) {
  __shared__ float sm[256];
  int b  = blockIdx.x;            // 0..4095 = c*64 + bz*8 + by
  int c  = b >> 6;
  int bz = (b >> 3) & 7;
  int by = b & 7;
  int t  = threadIdx.x;
  const float* base = x + (size_t)c * 2097152 + (size_t)bz * 16 * 16384 + (size_t)by * 2048;
  float acc = 0.f;
  for (int zi = 0; zi < 16; ++zi) {
    const float4* p = (const float4*)(base + (size_t)zi * 16384);
    float4 a = p[t];
    float4 q = p[t + 256];
    acc += a.x + a.y + a.z + a.w + q.x + q.y + q.z + q.w;
  }
  sm[t] = acc;
  __syncthreads();
  if (t < 8) {
    float s = 0.f;
    #pragma unroll
    for (int g = 0; g < 8; ++g)
      #pragma unroll
      for (int j = 0; j < 4; ++j)
        s += sm[g * 32 + t * 4 + j];
    s *= (1.0f / 4096.0f);
    seqH[c * DM + bz * 64 + by * 8 + t] = (_Float16)s;
  }
}

// ---------------------------------------------------------------------------
// Kernel 1: convert in_proj_w (2048x512) and out_proj_w (512x1024) to f16
// ---------------------------------------------------------------------------
__global__ void cvt_kernel(const float* __restrict__ w_in, const float* __restrict__ w_out,
                           _Float16* __restrict__ w_inH, _Float16* __restrict__ w_outH) {
  int stride = gridDim.x * blockDim.x;
  for (int i = blockIdx.x * blockDim.x + threadIdx.x; i < 1048576 + 524288; i += stride) {
    if (i < 1048576) w_inH[i] = (_Float16)w_in[i];
    else             w_outH[i - 1048576] = (_Float16)w_out[i - 1048576];
  }
}

// ---------------------------------------------------------------------------
// WMMA GEMM: C(MxN,f32) = A(MxK,f16) * Bw(NxK,f16)^T
// One wave per 16x16 tile; K in steps of 32 via v_wmma_f32_16x16x32_f16.
// A layout per ISA 16-bit 16x32 table: lane half selects K sub-blocks.
// B layout: lane = column, same K order as A (table order).
// ---------------------------------------------------------------------------
__global__ void wmma_gemm(const _Float16* __restrict__ A, const _Float16* __restrict__ Bw,
                          float* __restrict__ C, int lda, int ldb, int ldc, int K) {
  int mt = blockIdx.x, nt = blockIdx.y;
  int lane = threadIdx.x & 31;
  int hl = lane >> 4;             // lane half: 0 or 1
  int r  = lane & 15;
  const _Float16* Arow = A  + (size_t)(mt * 16 + r) * lda;
  const _Float16* Brow = Bw + (size_t)(nt * 16 + r) * ldb;
  v8f acc = {};
  for (int k0 = 0; k0 < K; k0 += 32) {
    h8 alo = *(const h8*)(Arow + k0 + hl * 8);        // K = 8*hl + 0..7
    h8 ahi = *(const h8*)(Arow + k0 + 16 + hl * 8);   // K = 16 + 8*hl + 0..7
    v16h a, bb;
    #pragma unroll
    for (int i = 0; i < 8; ++i) { a[i] = alo[i]; a[8 + i] = ahi[i]; }
    h8 blo = *(const h8*)(Brow + k0 + hl * 8);        // K = 8*hl + 0..7
    h8 bhi = *(const h8*)(Brow + k0 + 16 + hl * 8);   // K = 16 + 8*hl + 0..7
    #pragma unroll
    for (int i = 0; i < 8; ++i) { bb[i] = blo[i]; bb[8 + i] = bhi[i]; }
    acc = __builtin_amdgcn_wmma_f32_16x16x32_f16(false, a, false, bb,
                                                 (short)0, acc, false, false);
  }
  #pragma unroll
  for (int v = 0; v < 8; ++v)
    C[(size_t)(mt * 16 + v + 8 * hl) * ldc + nt * 16 + r] = acc[v];
}

// ---------------------------------------------------------------------------
// Kernel 3: causal depthwise conv1d (K=4) along L + bias + SiLU
// xz (64 x 2048, first 1024 cols = xs) -> xsC (64 x 1024)
// ---------------------------------------------------------------------------
__global__ void conv_silu_kernel(const float* __restrict__ xz, const float* __restrict__ conv_w,
                                 const float* __restrict__ conv_b, float* __restrict__ xsC) {
  int i = blockIdx.x * blockDim.x + threadIdx.x;  // 0..65535
  int l = i >> 10, e = i & (DI - 1);
  float s = conv_b[e];
  #pragma unroll
  for (int j = 0; j < 4; ++j) {
    int ls = l + j - 3;
    float xv = (ls >= 0) ? xz[(size_t)ls * (2 * DI) + e] : 0.f;
    s += conv_w[e * 4 + j] * xv;
  }
  xsC[i] = s / (1.f + __expf(-s));   // SiLU
}

// ---------------------------------------------------------------------------
// Kernel 4: x_dbl (64 x 64) = xsC (64x1024) @ x_proj_w(64x1024)^T
// ---------------------------------------------------------------------------
__global__ void xproj_kernel(const float* __restrict__ xsC, const float* __restrict__ xpw,
                             float* __restrict__ xdbl) {
  int i = blockIdx.x * blockDim.x + threadIdx.x;  // 0..4095
  int l = i >> 6, f = i & 63;
  const float* xr = xsC + (size_t)l * DI;
  const float* wr = xpw + (size_t)f * DI;
  float s = 0.f;
  for (int e = 0; e < DI; ++e) s += xr[e] * wr[e];
  xdbl[i] = s;
}

// ---------------------------------------------------------------------------
// Kernel 5: dt (64 x 1024) = softplus(x_dbl[:, :32] @ dt_proj_w^T + dt_proj_b)
// ---------------------------------------------------------------------------
__global__ void dt_kernel(const float* __restrict__ xdbl, const float* __restrict__ dtw,
                          const float* __restrict__ dtb, float* __restrict__ dtv) {
  int i = blockIdx.x * blockDim.x + threadIdx.x;  // 0..65535
  int l = i >> 10, e = i & (DI - 1);
  const float* dr = xdbl + (size_t)l * 64;
  const float* wr = dtw + (size_t)e * DTRANK;
  float s = dtb[e];
  #pragma unroll 8
  for (int r = 0; r < DTRANK; ++r) s += dr[r] * wr[r];
  dtv[i] = (s > 20.f) ? s : log1pf(__expf(s));    // softplus
}

// ---------------------------------------------------------------------------
// Kernel 6: selective scan over L=64, one thread per e, h[16] in registers.
// y = scan + D*xs, gated by silu(z); output f16 (64 x 1024) for out_proj.
// ---------------------------------------------------------------------------
__global__ void scan_kernel(const float* __restrict__ dtv, const float* __restrict__ xsC,
                            const float* __restrict__ xdbl, const float* __restrict__ xz,
                            const float* __restrict__ A_log, const float* __restrict__ Dp,
                            _Float16* __restrict__ yH) {
  int e = blockIdx.x * blockDim.x + threadIdx.x;  // 0..1023
  float As[DS], h[DS];
  #pragma unroll
  for (int s = 0; s < DS; ++s) { As[s] = -__expf(A_log[e * DS + s]); h[s] = 0.f; }
  float Dv = Dp[e];
  for (int l = 0; l < LSEQ; ++l) {
    float dt  = dtv[l * DI + e];
    float xv  = xsC[l * DI + e];
    float dtx = dt * xv;
    const float* Bs = xdbl + l * 64 + DTRANK;
    const float* Cs = xdbl + l * 64 + DTRANK + DS;
    float y = 0.f;
    #pragma unroll
    for (int s = 0; s < DS; ++s) {
      float da = __expf(dt * As[s]);
      h[s] = da * h[s] + dtx * Bs[s];
      y += h[s] * Cs[s];
    }
    y += Dv * xv;
    float zv = xz[l * (2 * DI) + DI + e];
    float g  = zv / (1.f + __expf(-zv));          // silu(z)
    yH[l * DI + e] = (_Float16)(y * g);
  }
}

// ---------------------------------------------------------------------------
extern "C" void kernel_launch(void* const* d_in, const int* in_sizes, int n_in,
                              void* d_out, int out_size, void* d_ws, size_t ws_size,
                              hipStream_t stream) {
  (void)in_sizes; (void)n_in; (void)out_size; (void)ws_size;
  const float* x      = (const float*)d_in[0];
  const float* in_w   = (const float*)d_in[1];
  const float* conv_w = (const float*)d_in[2];
  const float* conv_b = (const float*)d_in[3];
  const float* xp_w   = (const float*)d_in[4];
  const float* dt_w   = (const float*)d_in[5];
  const float* dt_b   = (const float*)d_in[6];
  const float* A_log  = (const float*)d_in[7];
  const float* Dp     = (const float*)d_in[8];
  const float* out_w  = (const float*)d_in[9];
  float* out = (float*)d_out;

  char* w = (char*)d_ws;
  _Float16* seqH  = (_Float16*)(w + 0);          //  64* 512 f16 =   64 KB
  _Float16* wInH  = (_Float16*)(w + 65536);      // 2048* 512 f16 = 2048 KB
  _Float16* wOutH = (_Float16*)(w + 2162688);    //  512*1024 f16 = 1024 KB
  float*    xzb   = (float*)   (w + 3211264);    //  64*2048 f32  =  512 KB
  float*    xsC   = (float*)   (w + 3735552);    //  64*1024 f32  =  256 KB
  float*    xdbl  = (float*)   (w + 3997696);    //  64*  64 f32  =   16 KB
  float*    dtv   = (float*)   (w + 4014080);    //  64*1024 f32  =  256 KB
  _Float16* yH    = (_Float16*)(w + 4276224);    //  64*1024 f16  =  128 KB

  // 1) bandwidth-bound pooling of 512 MB input
  pool_kernel<<<4096, 256, 0, stream>>>(x, seqH);
  // 2) one-time f16 weight conversion (3 MB write)
  cvt_kernel<<<1024, 256, 0, stream>>>(in_w, out_w, wInH, wOutH);
  // 3) in_proj: (64x512)@(2048x512)^T -> xz (64x2048), WMMA f16->f32
  wmma_gemm<<<dim3(4, 128), 32, 0, stream>>>(seqH, wInH, xzb, DM, DM, 2 * DI, DM);
  // 4) depthwise causal conv + SiLU
  conv_silu_kernel<<<256, 256, 0, stream>>>(xzb, conv_w, conv_b, xsC);
  // 5) x_proj -> (dt_low | B | C)
  xproj_kernel<<<16, 256, 0, stream>>>(xsC, xp_w, xdbl);
  // 6) dt_proj + softplus
  dt_kernel<<<256, 256, 0, stream>>>(xdbl, dt_w, dt_b, dtv);
  // 7) selective scan + D skip + SiLU(z) gate -> yH f16
  scan_kernel<<<4, 256, 0, stream>>>(dtv, xsC, xdbl, xzb, A_log, Dp, yH);
  // 8) out_proj: (64x1024)@(512x1024)^T -> out (64x512) = (1,64,8,8,8), WMMA
  wmma_gemm<<<dim3(4, 32), 32, 0, stream>>>(yH, wOutH, out, DI, DI, DM, DI);
}
